// LSHPerception_69028714381751
// MI455X (gfx1250) — matrix operand order repaired and checked by reading, not compiled, not run on hardware
//
#include <hip/hip_runtime.h>
#include <hip/hip_bf16.h>
#include <math.h>

// ---------------------------------------------------------------------------
// Reformer LSH attention for MI455X (gfx1250, wave32, WMMA).
// All matmul work (QK/V/O projections, QK^T dots, P@V) runs on
// v_wmma_f32_16x16x32_f16 (f16 in, f32 accumulate). Hashing, counting sort,
// masking, logsumexp and the hash-round softmax stay in f32 VALU.
// ---------------------------------------------------------------------------

typedef __attribute__((ext_vector_type(16))) _Float16 v16h;
typedef __attribute__((ext_vector_type(8)))  _Float16 v8h;
typedef __attribute__((ext_vector_type(8)))  float    v8f;

#define B_      8
#define S_      2048
#define D_      512
#define H_      8
#define DH      64
#define BH_     (B_ * H_)          // 64 batch-heads
#define NHASH   4
#define NBUCK   32                 // S_/64 buckets per hash
#define BS_     64                 // bucket (chunk) size
#define NCHUNK  (NHASH * NBUCK)    // 128 chunks per batch-head
#define TOTAL   (NHASH * S_)       // 8192 sorted slots per batch-head
#define MASKVAL (-1e9f)
#define SELFVAL (-5e4f)

static __device__ __forceinline__ v16h cat8(v8h lo, v8h hi) {
  return __builtin_shufflevector(lo, hi, 0,1,2,3,4,5,6,7,8,9,10,11,12,13,14,15);
}

// ---------------------------------------------------------------------------
// GEMM: out[M,N] = A[M,K] @ W[K,N] (+bias). A rows at stride lda (elements).
// Block = 128 threads (4 waves), block tile = 64(M) x 64(N). Wave w owns
// n-strip w and all 4 m-tiles: per 32-wide k-step it loads ONE B fragment and
// four A fragments -> 16 WMMAs per k-step per block. LDS tiles are stored in
// fragment order (B transposed) so every fragment is two contiguous v8h loads.
// ---------------------------------------------------------------------------
__global__ void __launch_bounds__(128)
gemm_wmma_kernel(const float* __restrict__ A, long lda,
                 const float* __restrict__ W,
                 const float* __restrict__ bias,
                 float* __restrict__ out,
                 int M, int N, int K) {
  __shared__ __align__(32) _Float16 a_lds[64 * 32];   // [m][kk]
  __shared__ __align__(32) _Float16 bT_lds[64 * 32];  // [n][kk]
  const int tid  = threadIdx.x;
  const int wave = tid >> 5, lane = tid & 31;
  const int lidx = lane & 15, khalf = lane >> 4;
  const int bm = blockIdx.x, bn = blockIdx.y;
  const bool full = (bm * 64 + 64) <= M;

  v8f zero = {};
  v8f acc[4];
#pragma unroll
  for (int mt = 0; mt < 4; ++mt) acc[mt] = zero;

  for (int kt = 0; kt < K / 32; ++kt) {
    if (full) {
      for (int e = tid; e < 64 * 32; e += 128) {
        int m = e >> 5, kk = e & 31;
        a_lds[e] = (_Float16)A[(long)(bm * 64 + m) * lda + kt * 32 + kk];
      }
    } else {
      for (int e = tid; e < 64 * 32; e += 128) {
        int m = e >> 5, kk = e & 31;
        int gm = bm * 64 + m;
        a_lds[e] = (gm < M) ? (_Float16)A[(long)gm * lda + kt * 32 + kk]
                            : (_Float16)0.f;
      }
    }
    for (int e = tid; e < 64 * 32; e += 128) {
      int n = e >> 5, kk = e & 31;
      bT_lds[e] = (_Float16)W[(long)(kt * 32 + kk) * N + bn * 64 + n];
    }
    __syncthreads();
    // B fragment: lane -> N = wave*16+lidx, K halves at khalf*8 / 16+khalf*8
    const _Float16* pb = &bT_lds[(wave * 16 + lidx) * 32 + khalf * 8];
    v16h bfrag = cat8(*(const v8h*)pb, *(const v8h*)(pb + 16));
#pragma unroll
    for (int mt = 0; mt < 4; ++mt) {
      // A fragment: lane -> M = mt*16+lidx, K halves at khalf*8 / 16+khalf*8
      const _Float16* pa = &a_lds[(mt * 16 + lidx) * 32 + khalf * 8];
      v16h afrag = cat8(*(const v8h*)pa, *(const v8h*)(pa + 16));
      acc[mt] = __builtin_amdgcn_wmma_f32_16x16x32_f16(false, afrag, false,
                                                       bfrag, (short)0,
                                                       acc[mt], false, false);
    }
    __syncthreads();
  }

  const int gn = bn * 64 + wave * 16 + lidx;
  const float bval = bias ? bias[gn] : 0.f;
#pragma unroll
  for (int mt = 0; mt < 4; ++mt) {
    int row0 = bm * 64 + mt * 16 + 8 * khalf;   // rows row0 .. row0+7
    float* op = out + (long)row0 * N + gn;
    if (full) {
#pragma unroll
      for (int r = 0; r < 8; ++r) { *op = acc[mt][r] + bval; op += N; }
    } else {
#pragma unroll
      for (int r = 0; r < 8; ++r) {
        if (row0 + r < M) *op = acc[mt][r] + bval;
        op += N;
      }
    }
  }
}

// ---------------------------------------------------------------------------
// LSH hashing: bucket[bh][h][s] = argmax over 32 of [rot, -rot].
// One wave per (bh, s); rotation table (16 KB) + q rows staged in LDS.
// Lane handles bins (h=lane>>4, r=lane&15) and (h+2, r); cross-lane argmax
// inside 16-lane groups via shfl_xor.
// ---------------------------------------------------------------------------
__global__ void __launch_bounds__(128)
hash_kernel(const float* __restrict__ qk,
            const float* __restrict__ rot,     // [DH][NHASH][16]
            int* __restrict__ buckets) {       // [BH][NHASH][S]
  __shared__ float rot_lds[DH * NHASH * 16];   // 16 KB
  __shared__ float q_lds[4][DH];
  const int warp = threadIdx.x >> 5, lane = threadIdx.x & 31;
  for (int e = threadIdx.x; e < DH * NHASH * 16; e += 128) rot_lds[e] = rot[e];
  int wid = blockIdx.x * 4 + warp;
  int bh = wid / S_, s = wid % S_;
  int b = bh >> 3, h = bh & 7;
  const float* q = qk + ((long)(b * S_ + s)) * D_ + h * DH;
  q_lds[warp][lane]      = q[lane];
  q_lds[warp][lane + 32] = q[lane + 32];
  __syncthreads();

  int r  = lane & 15;
  int h0 = lane >> 4;   // 0 or 1
  int h1 = h0 + 2;      // 2 or 3
  float a0 = 0.f, a1 = 0.f;
#pragma unroll 8
  for (int d = 0; d < DH; ++d) {
    float qv = q_lds[warp][d];
    a0 += qv * rot_lds[(d * NHASH + h0) * 16 + r];
    a1 += qv * rot_lds[(d * NHASH + h1) * 16 + r];
  }
  float v0 = (a0 >= -a0) ? a0 : -a0; int i0 = (a0 >= -a0) ? r : r + 16;
  float v1 = (a1 >= -a1) ? a1 : -a1; int i1 = (a1 >= -a1) ? r : r + 16;
  for (int m = 1; m < 16; m <<= 1) {
    float ov = __shfl_xor(v0, m); int oi = __shfl_xor(i0, m);
    if (ov > v0 || (ov == v0 && oi < i0)) { v0 = ov; i0 = oi; }
    ov = __shfl_xor(v1, m); oi = __shfl_xor(i1, m);
    if (ov > v1 || (ov == v1 && oi < i1)) { v1 = ov; i1 = oi; }
  }
  if (r == 0) {
    buckets[((long)bh * NHASH + h0) * S_ + s] = i0;
    buckets[((long)bh * NHASH + h1) * S_ + s] = i1;
  }
}

// ---------------------------------------------------------------------------
// Stable counting sort per (bh, hash): thread t owns bucket t.
// Matches jnp.argsort(bucket*S + pos) stability.
// ---------------------------------------------------------------------------
__global__ void __launch_bounds__(32)
sort_kernel(const int* __restrict__ buckets,
            int* __restrict__ sorted_pos,    // [BH][NHASH][S] -> original s
            int* __restrict__ unsort_idx) {  // [BH][NHASH][S] -> sorted slot
  int bhh = blockIdx.x;                      // bh*NHASH + h
  int t = threadIdx.x;                       // owns bucket t (0..31)
  const int* bk = buckets + (long)bhh * S_;
  int cnt = 0;
  for (int s = 0; s < S_; ++s) cnt += (bk[s] == t);
  int inc = cnt;
  for (int i = 1; i < 32; i <<= 1) {
    int v = __shfl_up(inc, i);
    if (t >= i) inc += v;
  }
  int pos = inc - cnt;                       // exclusive prefix = bucket base
  int* sp = sorted_pos + (long)bhh * S_;
  int* up = unsort_idx + (long)bhh * S_;
  for (int s = 0; s < S_; ++s) {
    if (bk[s] == t) { sp[pos] = s; up[s] = pos; ++pos; }
  }
}

// ---------------------------------------------------------------------------
// Chunked attention: one block (128 thr = 4 waves) per (bh, chunk).
// bq: 64x64 (raw), bk: 128x64 (L2-normalized, look-one-back), bv: 128x64.
// Wave w owns query rows 16w..16w+15: 16 WMMAs for dots, f32 mask/softmax,
// 16 WMMAs for P@V. Writes sorted outputs + per-row logsumexp.
// ---------------------------------------------------------------------------
__global__ void __launch_bounds__(128)
chunk_attn_kernel(const float* __restrict__ qk,
                  const float* __restrict__ v,
                  const unsigned char* __restrict__ mask,   // [B][S] bool
                  const int* __restrict__ sorted_pos,
                  float* __restrict__ so,                   // [BH][TOTAL][DH]
                  float* __restrict__ slog) {               // [BH][TOTAL]
  __shared__ __align__(32) _Float16 q_lds[64 * DH];     //  8 KB, row-major
  __shared__ __align__(32) _Float16 k_lds[128 * DH];    // 16 KB, row-major
  __shared__ __align__(32) _Float16 vT_lds[DH * 128];   // 16 KB, transposed
  __shared__ __align__(32) _Float16 p_lds[4][16 * 128]; // 16 KB, per wave
  __shared__ int pos_q[64];
  __shared__ int pos_k[128];
  __shared__ unsigned char mq[64];
  __shared__ unsigned char mk[128];

  const int bh = blockIdx.x / NCHUNK;
  const int c  = blockIdx.x % NCHUNK;
  const int tid = threadIdx.x;
  const int b = bh >> 3, h = bh & 7;
  const int* sp = sorted_pos + (long)bh * TOTAL;

  { // stage keys + values (look-one-back: rows 64..127 from chunk c-1 mod 128)
    int j = tid;
    int csrc = (j < BS_) ? c : (c + NCHUNK - 1) % NCHUNK;
    int s = sp[csrc * BS_ + (j & (BS_ - 1))];
    pos_k[j] = s;
    mk[j] = mask[b * S_ + s];
    const float* kr = qk + ((long)(b * S_ + s)) * D_ + h * DH;
    float ss = 0.f;
    for (int d0 = 0; d0 < DH; d0 += 4) {
      float4 f = *(const float4*)(kr + d0);
      ss += f.x * f.x + f.y * f.y + f.z * f.z + f.w * f.w;
    }
    float sc = 1.f / fmaxf(sqrtf(ss), 1e-9f);
    for (int d0 = 0; d0 < DH; d0 += 4) {
      float4 f = *(const float4*)(kr + d0);
      k_lds[j * DH + d0 + 0] = (_Float16)(f.x * sc);
      k_lds[j * DH + d0 + 1] = (_Float16)(f.y * sc);
      k_lds[j * DH + d0 + 2] = (_Float16)(f.z * sc);
      k_lds[j * DH + d0 + 3] = (_Float16)(f.w * sc);
    }
    const float* vr = v + ((long)(b * S_ + s)) * D_ + h * DH;
    for (int d = 0; d < DH; ++d) vT_lds[d * 128 + j] = (_Float16)vr[d];
  }
  if (tid < 64) { // stage queries (raw)
    int s = sp[c * BS_ + tid];
    pos_q[tid] = s;
    mq[tid] = mask[b * S_ + s];
    const float* qr = qk + ((long)(b * S_ + s)) * D_ + h * DH;
    for (int d = 0; d < DH; ++d) q_lds[tid * DH + d] = (_Float16)qr[d];
  }
  __syncthreads();

  const int wave = tid >> 5, lane = tid & 31;
  const int lidx = lane & 15, khalf = lane >> 4;
  const int m0 = wave * 16;

  // Q fragments (K = 64 -> 2 fragments of 16x32)
  v16h aq[2];
#pragma unroll
  for (int t = 0; t < 2; ++t) {
    const _Float16* p = &q_lds[(m0 + lidx) * DH + t * 32 + khalf * 8];
    aq[t] = cat8(*(const v8h*)p, *(const v8h*)(p + 16));
  }

  // dots = (q @ k^T) for 8 n-tiles of 16 keys each
  v8f dots[8];
#pragma unroll
  for (int j = 0; j < 8; ++j) {
    v8f acc = {};
#pragma unroll
    for (int t = 0; t < 2; ++t) {
      const _Float16* p = &k_lds[(j * 16 + lidx) * DH + t * 32 + khalf * 8];
      v16h bf = cat8(*(const v8h*)p, *(const v8h*)(p + 16));
      acc = __builtin_amdgcn_wmma_f32_16x16x32_f16(false, aq[t], false, bf,
                                                   (short)0, acc, false, false);
    }
    dots[j] = acc;
  }

  // scale + pad-mask + self-mask  (C layout: m = m0 + r + 8*khalf, n = j*16+lidx)
#pragma unroll
  for (int j = 0; j < 8; ++j) {
    int n = j * 16 + lidx;
    int pk = pos_k[n];
    unsigned char mkn = mk[n];
#pragma unroll
    for (int r = 0; r < 8; ++r) {
      int m = m0 + r + 8 * khalf;
      float dv = dots[j][r] * 0.125f;               // d^-0.5, d=64
      if (!(mq[m] && mkn)) dv = MASKVAL;
      if (pos_q[m] == pk)  dv = SELFVAL;            // applied after mask
      dots[j][r] = dv;
    }
  }

  // per-row logsumexp over 128 keys; probs -> per-wave LDS (single exp pass)
#pragma unroll
  for (int r = 0; r < 8; ++r) {
    float mx = dots[0][r];
#pragma unroll
    for (int j = 1; j < 8; ++j) mx = fmaxf(mx, dots[j][r]);
    for (int m = 1; m < 16; m <<= 1) mx = fmaxf(mx, __shfl_xor(mx, m));
    float se = 0.f;
#pragma unroll
    for (int j = 0; j < 8; ++j) {
      float e = expf(dots[j][r] - mx);
      dots[j][r] = e;
      se += e;
    }
    for (int m = 1; m < 16; m <<= 1) se += __shfl_xor(se, m);
    float inv = 1.f / se;                           // se >= exp(0) of max term
    float lse = mx + logf(se);
    int mloc = r + 8 * khalf;
#pragma unroll
    for (int j = 0; j < 8; ++j)
      p_lds[wave][mloc * 128 + j * 16 + lidx] = (_Float16)(dots[j][r] * inv);
    if (lidx == 0) slog[(long)bh * TOTAL + c * BS_ + m0 + mloc] = lse;
  }

  // bo = probs(16x128) @ v(128x64): 4 k-steps x 4 n-tiles
  v8f zero = {};
  v8f oacc[4];
#pragma unroll
  for (int jt = 0; jt < 4; ++jt) oacc[jt] = zero;
#pragma unroll
  for (int t = 0; t < 4; ++t) {
    const _Float16* pa = &p_lds[wave][lidx * 128 + t * 32 + khalf * 8];
    v16h af = cat8(*(const v8h*)pa, *(const v8h*)(pa + 16));
#pragma unroll
    for (int jt = 0; jt < 4; ++jt) {
      const _Float16* pb = &vT_lds[(jt * 16 + lidx) * 128 + t * 32 + khalf * 8];
      v16h bf = cat8(*(const v8h*)pb, *(const v8h*)(pb + 16));
      oacc[jt] = __builtin_amdgcn_wmma_f32_16x16x32_f16(false, af, false, bf,
                                                        (short)0, oacc[jt],
                                                        false, false);
    }
  }
  {
    long gbase = ((long)bh * TOTAL + c * BS_ + m0 + 8 * khalf) * DH + lidx;
#pragma unroll
    for (int jt = 0; jt < 4; ++jt) {
      float* op = so + gbase + jt * 16;
#pragma unroll
      for (int r = 0; r < 8; ++r) { *op = oacc[jt][r]; op += DH; }
    }
  }
}

// ---------------------------------------------------------------------------
// Unsort + softmax over hash rounds; write head-merged [B,S,D].
// ---------------------------------------------------------------------------
__global__ void __launch_bounds__(64)
combine_kernel(const float* __restrict__ so,
               const float* __restrict__ slog,
               const int* __restrict__ unsort_idx,
               float* __restrict__ merged) {
  int bs = blockIdx.x;                 // bh*S + s
  int bh = bs / S_, s = bs % S_;
  int t = threadIdx.x;                 // dim 0..63
  int b = bh >> 3, h = bh & 7;
  int u[NHASH];
  float l[NHASH];
  float mx = -3.0e38f;
#pragma unroll
  for (int hh = 0; hh < NHASH; ++hh) {
    u[hh] = unsort_idx[(long)bh * TOTAL + hh * S_ + s];
    l[hh] = slog[(long)bh * TOTAL + hh * S_ + u[hh]];
    mx = fmaxf(mx, l[hh]);
  }
  float se = 0.f;
#pragma unroll
  for (int hh = 0; hh < NHASH; ++hh) { l[hh] = expf(l[hh] - mx); se += l[hh]; }
  float inv = 1.f / se;
  float o = 0.f;
#pragma unroll
  for (int hh = 0; hh < NHASH; ++hh)
    o += l[hh] * inv * so[((long)bh * TOTAL + hh * S_ + u[hh]) * DH + t];
  merged[((long)(b * S_ + s)) * D_ + h * DH + t] = o;
}

// ---------------------------------------------------------------------------
// Decoder input: row 0 = curr_embedding, rows 1.. = encoder output C.
// Decoder mask: row 0 true, rest = memory_masks.
// ---------------------------------------------------------------------------
__global__ void __launch_bounds__(512)
build_dec_kernel(const float* __restrict__ enc_out,
                 const float* __restrict__ curr,
                 const unsigned char* __restrict__ in_mask,
                 float* __restrict__ dec_x,
                 unsigned char* __restrict__ dec_mask) {
  long bs = blockIdx.x;                // b*S + s
  int d = threadIdx.x;
  long b = bs / S_, s = bs % S_;
  float val = (s == 0) ? curr[b * D_ + d] : enc_out[bs * D_ + d];
  dec_x[bs * D_ + d] = val;
  if (d == 0) dec_mask[bs] = (s == 0) ? (unsigned char)1 : in_mask[bs];
}

// ---------------------------------------------------------------------------
extern "C" void kernel_launch(void* const* d_in, const int* in_sizes, int n_in,
                              void* d_out, int out_size, void* d_ws, size_t ws_size,
                              hipStream_t stream) {
  (void)in_sizes; (void)n_in; (void)out_size; (void)ws_size;
  const float* x        = (const float*)d_in[0];   // [8,2048,512]
  const float* curr     = (const float*)d_in[1];   // [8,1,512]
  const unsigned char* memmask = (const unsigned char*)d_in[2]; // [8,2048] bool
  const float* enc_Wqk  = (const float*)d_in[3];
  const float* enc_Wv   = (const float*)d_in[4];
  const float* enc_Wo   = (const float*)d_in[5];
  const float* enc_bo   = (const float*)d_in[6];
  const float* dec_Wqk  = (const float*)d_in[7];
  const float* dec_Wv   = (const float*)d_in[8];
  const float* dec_Wo   = (const float*)d_in[9];
  const float* dec_bo   = (const float*)d_in[10];
  const float* enc_rot  = (const float*)d_in[11];  // [64,4,16]
  const float* dec_rot  = (const float*)d_in[12];
  float* out = (float*)d_out;                      // [8,512]

  char* ws = (char*)d_ws;
  size_t off = 0;
  auto alloc = [&](size_t bytes) -> void* {
    void* p = ws + off;
    off += (bytes + 255) & ~(size_t)255;
    return p;
  };
  float* qk      = (float*)alloc(sizeof(float) * (size_t)B_ * S_ * D_);
  float* vv      = (float*)alloc(sizeof(float) * (size_t)B_ * S_ * D_);
  float* merged  = (float*)alloc(sizeof(float) * (size_t)B_ * S_ * D_);
  float* enc_out = (float*)alloc(sizeof(float) * (size_t)B_ * S_ * D_);
  float* dec_x   = (float*)alloc(sizeof(float) * (size_t)B_ * S_ * D_);
  int*   buckets = (int*)alloc(sizeof(int) * (size_t)BH_ * NHASH * S_);
  int*   sortedp = (int*)alloc(sizeof(int) * (size_t)BH_ * NHASH * S_);
  int*   unsortp = (int*)alloc(sizeof(int) * (size_t)BH_ * NHASH * S_);
  float* slog    = (float*)alloc(sizeof(float) * (size_t)BH_ * TOTAL);
  float* so      = (float*)alloc(sizeof(float) * (size_t)BH_ * TOTAL * DH);
  unsigned char* dmask = (unsigned char*)alloc((size_t)B_ * S_);

  const int M = B_ * S_;                 // 16384
  dim3 gGemm((M + 63) / 64, D_ / 64);    // 256 x 8
  auto phase = [&](const float* xin, const unsigned char* msk,
                   const float* Wqk, const float* Wv, const float* rot) {
    gemm_wmma_kernel<<<gGemm, 128, 0, stream>>>(xin, (long)D_, Wqk, nullptr, qk, M, D_, D_);
    gemm_wmma_kernel<<<gGemm, 128, 0, stream>>>(xin, (long)D_, Wv,  nullptr, vv, M, D_, D_);
    hash_kernel<<<BH_ * S_ / 4, 128, 0, stream>>>(qk, rot, buckets);
    sort_kernel<<<BH_ * NHASH, 32, 0, stream>>>(buckets, sortedp, unsortp);
    chunk_attn_kernel<<<BH_ * NCHUNK, 128, 0, stream>>>(qk, vv, msk, sortedp, so, slog);
    combine_kernel<<<BH_ * S_, 64, 0, stream>>>(so, slog, unsortp, merged);
  };

  // ---- encoder: LSH self-attention over memory ----
  phase(x, memmask, enc_Wqk, enc_Wv, enc_rot);
  gemm_wmma_kernel<<<gGemm, 128, 0, stream>>>(merged, (long)D_, enc_Wo, enc_bo,
                                              enc_out, M, D_, D_);

  // ---- decoder: curr embedding + encoded memory ----
  build_dec_kernel<<<B_ * S_, D_, 0, stream>>>(enc_out, curr, memmask, dec_x, dmask);
  phase(dec_x, dmask, dec_Wqk, dec_Wv, dec_rot);

  // final projection of row s==0 per batch: A rows at stride S*D
  gemm_wmma_kernel<<<dim3(1, D_ / 64), 128, 0, stream>>>(
      merged, (long)S_ * D_, dec_Wo, dec_bo, out, B_, D_, D_);
}
